// StylelizedTransformerBlock_54357106098898
// MI455X (gfx1250) — compile-verified, matrix operand order checked by reading, not run on hardware
//
#include <hip/hip_runtime.h>

// ---------------------------------------------------------------------------
// Types / WMMA helpers (CDNA5 gfx1250, wave32)
// ---------------------------------------------------------------------------
typedef __attribute__((ext_vector_type(16))) _Float16 v16h;
typedef __attribute__((ext_vector_type(8)))  float    v8f;

__device__ inline v8f wmma_f32_16x16x32(v16h a, v16h b, v8f c) {
  // (neg_a, A, neg_b, B, c_mod, C, reuse_a, reuse_b)
  return __builtin_amdgcn_wmma_f32_16x16x32_f16(false, a, false, b, (short)0, c,
                                                false, false);
}

// A-operand (16x32 f16) fragment load per the documented CDNA5 layout:
// lane L holds row (L&15); lanes 0-15 cover k in {0..7,16..23}, lanes 16-31
// cover k in {8..15,24..31}; within a lane, VGPR v holds half-pair
// k = 2v + (v>=4 ? 8 : 0) (+8 for upper lane half).  Source is row-major
// with leading dim `ld`; this same addressing also builds the B operand of
// X @ W^T from row-major W (lane -> output column == W row).
__device__ inline v16h load_frag_rowpat(const _Float16* __restrict__ base,
                                        int row0, int ld, int k0) {
  int lane = threadIdx.x & 31;
  const _Float16* p = base + (long)(row0 + (lane & 15)) * ld + k0 + ((lane >> 4) << 3);
  v16h f;
#pragma unroll
  for (int v = 0; v < 8; ++v) {
    int k = 2 * v + (v >= 4 ? 8 : 0);
    f[2 * v]     = p[k];
    f[2 * v + 1] = p[k + 1];
  }
  return f;
}

// B-operand (32x16) fragment where B[k][n] = M[(k0+k)*ld + col0+n]
// (column-pattern access into a row-major K x Ncols matrix, e.g. V in P@V).
__device__ inline v16h load_frag_colpat(const _Float16* __restrict__ M,
                                        int k0, int col0, int ld) {
  int lane = threadIdx.x & 31;
  int col  = col0 + (lane & 15);
  int kh   = (lane >> 4) << 3;
  v16h f;
#pragma unroll
  for (int v = 0; v < 8; ++v) {
    int k = k0 + kh + 2 * v + (v >= 4 ? 8 : 0);
    f[2 * v]     = M[(long)k * ld + col];
    f[2 * v + 1] = M[(long)(k + 1) * ld + col];
  }
  return f;
}

// ---------------------------------------------------------------------------
// Problem constants
// ---------------------------------------------------------------------------
#define BB   8
#define NN   1024
#define FEAT 512
#define HID  512
#define FFN  1024
#define SDIM 512
#define HD   64
#define NH   8
#define MROWS (BB * NN)   // 8192

// ---------------------------------------------------------------------------
// Small fp32 VALU kernels: style projection, demodulation, casts, modulation
// ---------------------------------------------------------------------------
__global__ __launch_bounds__(256) void style_kernel(
    const float* __restrict__ s, const float* __restrict__ aW,
    const float* __restrict__ ab, float* __restrict__ style, int dim) {
  int idx = blockIdx.x * blockDim.x + threadIdx.x;
  if (idx >= BB * dim) return;
  int b = idx / dim, i = idx % dim;
  const float* sp = s + (long)b * SDIM;
  const float* wp = aW + (long)i * SDIM;
  float acc = 0.f;
  for (int j = 0; j < SDIM; ++j) acc += sp[j] * wp[j];
  style[idx] = acc + ab[i];
}

__global__ __launch_bounds__(256) void demod_kernel(
    const float* __restrict__ W, const float* __restrict__ style,
    float* __restrict__ demod, int out_dim, int in_dim) {
  int idx = blockIdx.x * blockDim.x + threadIdx.x;
  if (idx >= BB * out_dim) return;
  int b = idx / out_dim, o = idx % out_dim;
  const float* wp = W + (long)o * in_dim;
  const float* sp = style + (long)b * in_dim;
  float acc = 0.f;
  for (int i = 0; i < in_dim; ++i) {
    float t = wp[i] * sp[i];
    acc += t * t;
  }
  demod[idx] = rsqrtf(acc + 1e-8f);
}

__global__ __launch_bounds__(256) void cast_f16_kernel(
    const float* __restrict__ W, _Float16* __restrict__ Wh, long n) {
  long i = (long)blockIdx.x * blockDim.x + threadIdx.x;
  if (i >= n) return;
  Wh[i] = (_Float16)W[i];
}

// xs[b,n,i] = f16( X[b,n,i] * style[b,i] )
__global__ __launch_bounds__(256) void xmod_kernel(
    const float* __restrict__ X, const float* __restrict__ style,
    _Float16* __restrict__ Xs, int dim) {
  long i = (long)blockIdx.x * blockDim.x + threadIdx.x;
  long total = (long)MROWS * dim;
  if (i >= total) return;
  int b = (int)(i / ((long)NN * dim));
  int c = (int)(i % dim);
  Xs[i] = (_Float16)(X[i] * style[(long)b * dim + c]);
}

// ---------------------------------------------------------------------------
// Modulated GEMM:  Y[row, o] = act( demod[b,o] * sum_k Xs[row,k] * Wh[o,k] )
// Wave-level tile: 32 rows x 64 cols (2x4 WMMA accumulators).
// ---------------------------------------------------------------------------
__global__ __launch_bounds__(256) void modgemm_kernel(
    const _Float16* __restrict__ Xs, const _Float16* __restrict__ Wh,
    const float* __restrict__ demod, float* __restrict__ Yf,
    _Float16* __restrict__ Yh, int Kdim, int Odim, int relu) {
  int wid = (blockIdx.x * blockDim.x + threadIdx.x) >> 5;
  int numColG = Odim >> 6;             // 64-wide column groups
  int numRowT = MROWS >> 5;            // 32-row tiles
  if (wid >= numRowT * numColG) return;
  int rt = wid / numColG, cg = wid % numColG;
  int row0 = rt << 5;
  int col0 = cg << 6;
  int b = row0 >> 10;                  // 1024 rows per batch, tile never crosses

  v8f zero = {};
  v8f acc[2][4];
#pragma unroll
  for (int r = 0; r < 2; ++r)
#pragma unroll
    for (int c = 0; c < 4; ++c) acc[r][c] = zero;

  for (int k0 = 0; k0 < Kdim; k0 += 32) {
    v16h a0 = load_frag_rowpat(Xs, row0,      Kdim, k0);
    v16h a1 = load_frag_rowpat(Xs, row0 + 16, Kdim, k0);
#pragma unroll
    for (int sidx = 0; sidx < 4; ++sidx) {
      v16h bf = load_frag_rowpat(Wh, col0 + sidx * 16, Kdim, k0);
      acc[0][sidx] = wmma_f32_16x16x32(a0, bf, acc[0][sidx]);
      acc[1][sidx] = wmma_f32_16x16x32(a1, bf, acc[1][sidx]);
    }
  }

  int lane = threadIdx.x & 31;
  int n = lane & 15, hi = lane >> 4;
#pragma unroll
  for (int rs = 0; rs < 2; ++rs) {
#pragma unroll
    for (int sidx = 0; sidx < 4; ++sidx) {
      int col = col0 + sidx * 16 + n;
      float dm = demod[(long)b * Odim + col];
#pragma unroll
      for (int v = 0; v < 8; ++v) {
        int row = row0 + rs * 16 + v + 8 * hi;
        float y = acc[rs][sidx][v] * dm;
        if (relu) y = fmaxf(y, 0.f);
        long idx = (long)row * Odim + col;
        if (Yf) Yf[idx] = y;
        if (Yh) Yh[idx] = (_Float16)y;
      }
    }
  }
}

// ---------------------------------------------------------------------------
// Flash-style self-attention (q=k=v=h), one wave per (batch, head, 16-q tile).
// S = (Q K^T)/8 via WMMA, online softmax, O += P V via WMMA.  P converted
// from C-layout to A-layout through a per-wave LDS tile.
// ---------------------------------------------------------------------------
__global__ __launch_bounds__(256) void attn_kernel(
    const _Float16* __restrict__ H, float* __restrict__ O) {
  __shared__ _Float16 sm[8][16 * 34];           // per-wave 16x32 tile, pad 2
  int wvb  = threadIdx.x >> 5;
  int wid  = blockIdx.x * 8 + wvb;              // 4096 waves total, exact grid
  int qt   = wid & 63;
  int head = (wid >> 6) & 7;
  int b    = wid >> 9;
  int lane = threadIdx.x & 31;
  int hi = lane >> 4, n = lane & 15;

  const _Float16* Hb = H + (long)b * NN * HID;
  int q0 = qt << 4;
  int hc = head * HD;

  v16h qa0 = load_frag_rowpat(Hb, q0, HID, hc);
  v16h qa1 = load_frag_rowpat(Hb, q0, HID, hc + 32);

  v8f zero = {};
  v8f oacc[4];
#pragma unroll
  for (int sidx = 0; sidx < 4; ++sidx) oacc[sidx] = zero;

  float mrow[8], lrow[8];
#pragma unroll
  for (int v = 0; v < 8; ++v) { mrow[v] = -1e30f; lrow[v] = 0.f; }

  _Float16* smp = &sm[wvb][0];
  const float scale = 0.125f;                   // 1/sqrt(64)

  for (int j0 = 0; j0 < NN; j0 += 32) {
    // ---- S tiles: rows = queries, cols = 32 keys ----
    v16h kb00 = load_frag_rowpat(Hb, j0,      HID, hc);
    v16h kb01 = load_frag_rowpat(Hb, j0,      HID, hc + 32);
    v16h kb10 = load_frag_rowpat(Hb, j0 + 16, HID, hc);
    v16h kb11 = load_frag_rowpat(Hb, j0 + 16, HID, hc + 32);
    v8f s0 = zero, s1 = zero;
    s0 = wmma_f32_16x16x32(qa0, kb00, s0);
    s0 = wmma_f32_16x16x32(qa1, kb01, s0);
    s1 = wmma_f32_16x16x32(qa0, kb10, s1);
    s1 = wmma_f32_16x16x32(qa1, kb11, s1);

    // ---- online softmax over this 32-key block ----
    float tmax[8];
#pragma unroll
    for (int v = 0; v < 8; ++v) {
      s0[v] *= scale; s1[v] *= scale;
      tmax[v] = fmaxf(s0[v], s1[v]);
    }
#pragma unroll
    for (int msk = 1; msk < 16; msk <<= 1)
#pragma unroll
      for (int v = 0; v < 8; ++v)
        tmax[v] = fmaxf(tmax[v], __shfl_xor(tmax[v], msk, 32));

    float alpha[8];
#pragma unroll
    for (int v = 0; v < 8; ++v) {
      float mn = fmaxf(mrow[v], tmax[v]);
      alpha[v] = __expf(mrow[v] - mn);
      mrow[v] = mn;
    }
    float rs[8];
#pragma unroll
    for (int v = 0; v < 8; ++v) {
      s0[v] = __expf(s0[v] - mrow[v]);
      s1[v] = __expf(s1[v] - mrow[v]);
      rs[v] = s0[v] + s1[v];
    }
#pragma unroll
    for (int msk = 1; msk < 16; msk <<= 1)
#pragma unroll
      for (int v = 0; v < 8; ++v)
        rs[v] += __shfl_xor(rs[v], msk, 32);
#pragma unroll
    for (int v = 0; v < 8; ++v) {
      lrow[v] = lrow[v] * alpha[v] + rs[v];
      oacc[0][v] *= alpha[v]; oacc[1][v] *= alpha[v];
      oacc[2][v] *= alpha[v]; oacc[3][v] *= alpha[v];
    }

    // ---- C-layout -> A-layout via LDS (per-wave region, same-wave order) ----
#pragma unroll
    for (int v = 0; v < 8; ++v) {
      int m = v + 8 * hi;
      smp[m * 34 + n]      = (_Float16)s0[v];
      smp[m * 34 + 16 + n] = (_Float16)s1[v];
    }
    asm volatile("s_wait_dscnt 0x0" ::: "memory");
    v16h pf = load_frag_rowpat(smp, 0, 34, 0);

    // ---- O += P @ V ----
#pragma unroll
    for (int sidx = 0; sidx < 4; ++sidx) {
      v16h vb = load_frag_colpat(Hb, j0, hc + sidx * 16, HID);
      oacc[sidx] = wmma_f32_16x16x32(pf, vb, oacc[sidx]);
    }
  }

  // ---- epilogue: O / l ----
#pragma unroll
  for (int sidx = 0; sidx < 4; ++sidx) {
#pragma unroll
    for (int v = 0; v < 8; ++v) {
      int m = v + 8 * hi;
      long idx = ((long)b * NN + q0 + m) * HID + hc + sidx * 16 + n;
      O[idx] = oacc[sidx][v] / lrow[v];
    }
  }
}

// ---------------------------------------------------------------------------
// LayerNorm (optionally of a sum): out = LN(A (+ Bv)), one wave per row.
// ---------------------------------------------------------------------------
template <int DIM>
__global__ __launch_bounds__(256) void ln_kernel(
    const float* __restrict__ A, const float* __restrict__ Bv,
    float* __restrict__ out) {
  int wid  = (blockIdx.x * blockDim.x + threadIdx.x) >> 5;
  int lane = threadIdx.x & 31;
  if (wid >= MROWS) return;
  const float* a = A + (long)wid * DIM;
  constexpr int PER = DIM / 32;
  float vals[PER];
  float sum = 0.f;
#pragma unroll
  for (int t = 0; t < PER; ++t) {
    float v = a[lane + t * 32];
    if (Bv) v += Bv[(long)wid * DIM + lane + t * 32];
    vals[t] = v;
    sum += v;
  }
#pragma unroll
  for (int msk = 1; msk < 32; msk <<= 1) sum += __shfl_xor(sum, msk, 32);
  float mean = sum * (1.0f / DIM);
  float var = 0.f;
#pragma unroll
  for (int t = 0; t < PER; ++t) {
    float d = vals[t] - mean;
    var += d * d;
  }
#pragma unroll
  for (int msk = 1; msk < 32; msk <<= 1) var += __shfl_xor(var, msk, 32);
  float inv = rsqrtf(var * (1.0f / DIM) + 1e-5f);
#pragma unroll
  for (int t = 0; t < PER; ++t)
    out[(long)wid * DIM + lane + t * 32] = (vals[t] - mean) * inv;
}

// ---------------------------------------------------------------------------
// Host-side pipeline
// ---------------------------------------------------------------------------
extern "C" void kernel_launch(void* const* d_in, const int* in_sizes, int n_in,
                              void* d_out, int out_size, void* d_ws,
                              size_t ws_size, hipStream_t stream) {
  (void)in_sizes; (void)n_in; (void)out_size; (void)ws_size;
  const float* x   = (const float*)d_in[0];
  const float* s   = (const float*)d_in[1];
  const float* Wk  = (const float*)d_in[2];
  const float* aWk = (const float*)d_in[3];
  const float* abk = (const float*)d_in[4];
  const float* Wo  = (const float*)d_in[5];
  const float* aWo = (const float*)d_in[6];
  const float* abo = (const float*)d_in[7];
  const float* W1  = (const float*)d_in[8];
  const float* aW1 = (const float*)d_in[9];
  const float* ab1 = (const float*)d_in[10];
  const float* W2  = (const float*)d_in[11];
  const float* aW2 = (const float*)d_in[12];
  const float* ab2 = (const float*)d_in[13];
  float* out = (float*)d_out;

  char* ws = (char*)d_ws;
  size_t off = 0;
  auto alloc = [&](size_t bytes) -> char* {
    char* p = ws + off;
    off = (off + bytes + 255) & ~(size_t)255;
    return p;
  };

  float* style_k = (float*)alloc((size_t)BB * FEAT * 4);
  float* style_o = (float*)alloc((size_t)BB * HID * 4);
  float* style_1 = (float*)alloc((size_t)BB * FEAT * 4);
  float* style_2 = (float*)alloc((size_t)BB * FFN * 4);
  float* demod_k = (float*)alloc((size_t)BB * HID * 4);
  float* demod_o = (float*)alloc((size_t)BB * FEAT * 4);
  float* demod_1 = (float*)alloc((size_t)BB * FFN * 4);
  float* demod_2 = (float*)alloc((size_t)BB * FEAT * 4);
  _Float16* whk = (_Float16*)alloc((size_t)HID * FEAT * 2);
  _Float16* who = (_Float16*)alloc((size_t)FEAT * HID * 2);
  _Float16* wh1 = (_Float16*)alloc((size_t)FFN * FEAT * 2);
  _Float16* wh2 = (_Float16*)alloc((size_t)FEAT * FFN * 2);
  _Float16* xs  = (_Float16*)alloc((size_t)MROWS * FFN * 2);   // max-width input
  _Float16* h16 = (_Float16*)alloc((size_t)MROWS * HID * 2);
  float* bufO = (float*)alloc((size_t)MROWS * HID * 4);        // attn out, later f2
  float* bufT = (float*)alloc((size_t)MROWS * HID * 4);        // Wo out
  float* x2   = (float*)alloc((size_t)MROWS * FEAT * 4);       // ln(x + o)
  float* f1   = (float*)alloc((size_t)MROWS * FFN * 4);        // FFN hidden

  auto cdiv = [](long a, long b) { return (int)((a + b - 1) / b); };

  // --- styles ---
  style_kernel<<<cdiv(BB * FEAT, 256), 256, 0, stream>>>(s, aWk, abk, style_k, FEAT);
  style_kernel<<<cdiv(BB * HID,  256), 256, 0, stream>>>(s, aWo, abo, style_o, HID);
  style_kernel<<<cdiv(BB * FEAT, 256), 256, 0, stream>>>(s, aW1, ab1, style_1, FEAT);
  style_kernel<<<cdiv(BB * FFN,  256), 256, 0, stream>>>(s, aW2, ab2, style_2, FFN);

  // --- demodulation factors ---
  demod_kernel<<<cdiv(BB * HID,  256), 256, 0, stream>>>(Wk, style_k, demod_k, HID, FEAT);
  demod_kernel<<<cdiv(BB * FEAT, 256), 256, 0, stream>>>(Wo, style_o, demod_o, FEAT, HID);
  demod_kernel<<<cdiv(BB * FFN,  256), 256, 0, stream>>>(W1, style_1, demod_1, FFN, FEAT);
  demod_kernel<<<cdiv(BB * FEAT, 256), 256, 0, stream>>>(W2, style_2, demod_2, FEAT, FFN);

  // --- f16 weight casts ---
  cast_f16_kernel<<<cdiv((long)HID * FEAT, 256), 256, 0, stream>>>(Wk, whk, (long)HID * FEAT);
  cast_f16_kernel<<<cdiv((long)FEAT * HID, 256), 256, 0, stream>>>(Wo, who, (long)FEAT * HID);
  cast_f16_kernel<<<cdiv((long)FFN * FEAT, 256), 256, 0, stream>>>(W1, wh1, (long)FFN * FEAT);
  cast_f16_kernel<<<cdiv((long)FEAT * FFN, 256), 256, 0, stream>>>(W2, wh2, (long)FEAT * FFN);

  // --- h = modlinear(x; Wk) -> f16 for attention ---
  xmod_kernel<<<cdiv((long)MROWS * FEAT, 256), 256, 0, stream>>>(x, style_k, xs, FEAT);
  modgemm_kernel<<<(MROWS / 32) * (HID / 64) / 8, 256, 0, stream>>>(
      xs, whk, demod_k, nullptr, h16, FEAT, HID, 0);

  // --- attention (shared q=k=v=h) ---
  attn_kernel<<<(BB * NH * (NN / 16)) / 8, 256, 0, stream>>>(h16, bufO);

  // --- o = modlinear(attn; Wo) ---
  xmod_kernel<<<cdiv((long)MROWS * HID, 256), 256, 0, stream>>>(bufO, style_o, xs, HID);
  modgemm_kernel<<<(MROWS / 32) * (FEAT / 64) / 8, 256, 0, stream>>>(
      xs, who, demod_o, bufT, nullptr, HID, FEAT, 0);

  // --- x2 = LN(x + o) ---
  ln_kernel<FEAT><<<MROWS / 8, 256, 0, stream>>>(x, bufT, x2);

  // --- f1 = relu(modlinear(x2; W1)); f1 = LN(f1) ---
  xmod_kernel<<<cdiv((long)MROWS * FEAT, 256), 256, 0, stream>>>(x2, style_1, xs, FEAT);
  modgemm_kernel<<<(MROWS / 32) * (FFN / 64) / 8, 256, 0, stream>>>(
      xs, wh1, demod_1, f1, nullptr, FEAT, FFN, 1);
  ln_kernel<FFN><<<MROWS / 8, 256, 0, stream>>>(f1, nullptr, f1);

  // --- f2 = modlinear(f1; W2); f2 = LN(f2) ---
  xmod_kernel<<<cdiv((long)MROWS * FFN, 256), 256, 0, stream>>>(f1, style_2, xs, FFN);
  modgemm_kernel<<<(MROWS / 32) * (FEAT / 64) / 8, 256, 0, stream>>>(
      xs, wh2, demod_2, bufO, nullptr, FFN, FEAT, 0);
  ln_kernel<FEAT><<<MROWS / 8, 256, 0, stream>>>(bufO, nullptr, bufO);

  // --- out = LN(x2 + f2) ---
  ln_kernel<FEAT><<<MROWS / 8, 256, 0, stream>>>(x2, bufO, out);
}
